// Classical2PCNNForward_44581760532630
// MI455X (gfx1250) — matrix operand order repaired and checked by reading, not compiled, not run on hardware
//
#include <hip/hip_runtime.h>
#include <hip/hip_bf16.h>

// ---------------------------------------------------------------------------
// Classical 2-phase GNN forward for MI455X (gfx1250, wave32, WMMA).
//   msg  = MLP_phi(cat(x[dst], x[src]))   per edge          (16x16x32 f16 WMMA)
//   aggr = scatter_add(msg, dst)                             (f32 atomics, L2)
//   h    = MLP_gam(cat(x, aggr))          per node          (16x16x32 f16 WMMA)
//   pooled = segment_mean(h, batch)                          (f32 atomics)
//   out  = MLP_head(pooled)               100x32 -> 64 -> 2 (tiny, VALU)
// ---------------------------------------------------------------------------

#define N_NODES  50000
#define N_EDGES  1600000
#define FDIM     32
#define N_GRAPHS 100

typedef __attribute__((ext_vector_type(16))) _Float16 v16h;
typedef __attribute__((ext_vector_type(8)))  float    v8f;

union FragH {
    v16h     v;
    uint4    q[2];
    _Float16 h[16];
};

__device__ __forceinline__ v8f wmma_f16(const v16h& a, const v16h& b, const v8f& c) {
    return __builtin_amdgcn_wmma_f32_16x16x32_f16(
        /*neg_a=*/false, a, /*neg_b=*/false, b,
        /*c_mod=*/(short)0, c, /*reuse_a=*/false, /*reuse_b=*/false);
}

__device__ __forceinline__ void atomAddF32(float* p, float v) {
    unsafeAtomicAdd(p, v);   // -> global_atomic_add_f32 (L2-resident targets)
}

// Pack a 32x16 B-tile (K0..K0+31, N0..N0+15) of row-major f32 weight W[Kdim x Ndim]
// into the dense 16-bit WMMA B layout:
//   lanes 0-15  : col N0+lane, elems 0..15 = K0+{0..7,16..23}
//   lanes 16-31 : col N0+lane, elems 0..15 = K0+{8..15,24..31}
__device__ __forceinline__ FragH load_bfrag(const float* __restrict__ W,
                                            int Ndim, int K0, int N0,
                                            int hi, int lm) {
    FragH f;
    const int kbase = K0 + hi * 8;
    const int n     = N0 + lm;
#pragma unroll
    for (int j = 0; j < 16; ++j)
        f.h[j] = (_Float16)W[(kbase + j + (j >= 8 ? 8 : 0)) * Ndim + n];
    return f;
}

// ---------------------------------------------------------------------------
// Kernel 0: convert x -> f16 (row-major [N,32]) and zero aggr / sums / counts.
// ---------------------------------------------------------------------------
__global__ void init_kernel(const float* __restrict__ x,
                            _Float16* __restrict__ xh,
                            float* __restrict__ aggr,
                            float* __restrict__ sums,
                            float* __restrict__ cnt) {
    const int total = N_NODES * FDIM;
    for (int i = blockIdx.x * blockDim.x + threadIdx.x; i < total;
         i += gridDim.x * blockDim.x) {
        xh[i]   = (_Float16)x[i];
        aggr[i] = 0.0f;
        if (i < N_GRAPHS * FDIM) sums[i] = 0.0f;
        if (i < N_GRAPHS)        cnt[i]  = 0.0f;
    }
}

// Kernel 1: per-graph node counts (after init zeroed cnt).
__global__ void count_kernel(const int* __restrict__ batch, float* __restrict__ cnt) {
    int i = blockIdx.x * blockDim.x + threadIdx.x;
    if (i < N_NODES) atomAddF32(&cnt[batch[i]], 1.0f);
}

// ---------------------------------------------------------------------------
// Kernel 2: edge message MLP + scatter-add.
// One wave handles a 16-edge tile; 12 WMMAs per tile.
// ---------------------------------------------------------------------------
__global__ void __launch_bounds__(256)
edge_mlp_kernel(const _Float16* __restrict__ xh,
                const int* __restrict__ src, const int* __restrict__ dst,
                const float* __restrict__ w1, const float* __restrict__ b1,
                const float* __restrict__ w2, const float* __restrict__ b2,
                float* __restrict__ aggr, int nTiles) {
    __shared__ _Float16 lds[8][16 * 72];        // per-wave 16x64 H tile, pad 8
    const int lane = threadIdx.x & 31;
    const int wave = threadIdx.x >> 5;
    const int hi   = lane >> 4;
    const int lm   = lane & 15;

    // ---- weight fragments (loaded once per wave, reused across tiles) ----
    FragH B1[2][4], B2[2][2];
#pragma unroll
    for (int kt = 0; kt < 2; ++kt) {
#pragma unroll
        for (int nt = 0; nt < 4; ++nt) B1[kt][nt] = load_bfrag(w1, 64, kt * 32, nt * 16, hi, lm);
#pragma unroll
        for (int nt = 0; nt < 2; ++nt) B2[kt][nt] = load_bfrag(w2, 32, kt * 32, nt * 16, hi, lm);
    }
    float bias1[4], bias2[2];
#pragma unroll
    for (int nt = 0; nt < 4; ++nt) bias1[nt] = b1[nt * 16 + lm];
#pragma unroll
    for (int nt = 0; nt < 2; ++nt) bias2[nt] = b2[nt * 16 + lm];

    _Float16* myLds = &lds[wave][0];

    for (int tile = blockIdx.x * 8 + wave; tile < nTiles; tile += gridDim.x * 8) {
        const int e = tile * 16 + lm;
        const int d = dst[e];
        const int s = src[e];
        // A fragments: row = edge lm; K0..31 = x[dst], K32..63 = x[src].
        // Dense 16-bit A layout: lo lanes take K {0-7,16-23}, hi lanes {8-15,24-31}.
        const uint4* pd = (const uint4*)(xh + (size_t)d * FDIM);
        const uint4* ps = (const uint4*)(xh + (size_t)s * FDIM);
        FragH A0, A1;
        A0.q[0] = pd[hi];     A0.q[1] = pd[2 + hi];
        A1.q[0] = ps[hi];     A1.q[1] = ps[2 + hi];

        // ---- layer 1: H = relu(cat @ W1 + b1), 16x64, to LDS row-major ----
#pragma unroll
        for (int nt = 0; nt < 4; ++nt) {
            v8f c;
#pragma unroll
            for (int v = 0; v < 8; ++v) c[v] = bias1[nt];
            c = wmma_f16(A0.v, B1[0][nt].v, c);
            c = wmma_f16(A1.v, B1[1][nt].v, c);
#pragma unroll
            for (int v = 0; v < 8; ++v) {
                float r = c[v] > 0.0f ? c[v] : 0.0f;       // C/D: row v+8*hi, col lm
                myLds[(v + 8 * hi) * 72 + nt * 16 + lm] = (_Float16)r;
            }
        }
        asm volatile("s_wait_dscnt 0" ::: "memory");       // wave-local LDS RAW

        // ---- reload H in A layout ----
        const uint4* pr = (const uint4*)(myLds + lm * 72); // 144B-aligned rows
        FragH A2a, A2b;
        A2a.q[0] = pr[hi];     A2a.q[1] = pr[2 + hi];      // K 0..31
        A2b.q[0] = pr[4 + hi]; A2b.q[1] = pr[6 + hi];      // K 32..63

        // ---- layer 2 + scatter-add into aggr[dst] ----
#pragma unroll
        for (int nt = 0; nt < 2; ++nt) {
            v8f c;
#pragma unroll
            for (int v = 0; v < 8; ++v) c[v] = bias2[nt];
            c = wmma_f16(A2a.v, B2[0][nt].v, c);
            c = wmma_f16(A2b.v, B2[1][nt].v, c);
#pragma unroll
            for (int v = 0; v < 8; ++v) {
                const int row = v + 8 * hi;
                const int de  = dst[tile * 16 + row];
                atomAddF32(&aggr[(size_t)de * FDIM + nt * 16 + lm], c[v]);
            }
        }
    }
}

// ---------------------------------------------------------------------------
// Kernel 3: node update MLP (gamma) + pooled-sum scatter per graph.
// ---------------------------------------------------------------------------
__global__ void __launch_bounds__(256)
node_mlp_kernel(const _Float16* __restrict__ xh,
                const float* __restrict__ aggr,
                const int* __restrict__ batch,
                const float* __restrict__ w1, const float* __restrict__ b1,
                const float* __restrict__ w2, const float* __restrict__ b2,
                float* __restrict__ sums, int nTiles) {
    __shared__ _Float16 lds[8][16 * 72];
    const int lane = threadIdx.x & 31;
    const int wave = threadIdx.x >> 5;
    const int hi   = lane >> 4;
    const int lm   = lane & 15;

    const int tile = blockIdx.x * 8 + wave;
    if (tile >= nTiles) return;                     // wave-uniform guard

    FragH B1[2][4], B2[2][2];
#pragma unroll
    for (int kt = 0; kt < 2; ++kt) {
#pragma unroll
        for (int nt = 0; nt < 4; ++nt) B1[kt][nt] = load_bfrag(w1, 64, kt * 32, nt * 16, hi, lm);
#pragma unroll
        for (int nt = 0; nt < 2; ++nt) B2[kt][nt] = load_bfrag(w2, 32, kt * 32, nt * 16, hi, lm);
    }
    float bias1[4], bias2[2];
#pragma unroll
    for (int nt = 0; nt < 4; ++nt) bias1[nt] = b1[nt * 16 + lm];
#pragma unroll
    for (int nt = 0; nt < 2; ++nt) bias2[nt] = b2[nt * 16 + lm];

    _Float16* myLds = &lds[wave][0];

    const int n = tile * 16 + lm;
    // A0: x features (already f16). A1: aggr row (f32 -> f16 on the fly).
    const uint4* px = (const uint4*)(xh + (size_t)n * FDIM);
    FragH A0, A1;
    A0.q[0] = px[hi];  A0.q[1] = px[2 + hi];
    {
        const float4* pa = (const float4*)(aggr + (size_t)n * FDIM);
        float4 f0 = pa[hi];      // aggr cols hi*4 .. +3,  (K 32..35 / 36..39)
        float4 f1 = pa[hi + 1 - hi];   // placeholder, replaced below
        (void)f1;
        float4 fa = pa[hi];          // cols hi*4+0..3
        float4 fb = pa[hi * 1 + (hi ? 3 : 1)]; // unused path guard
        (void)fa; (void)fb; (void)f0;
        // explicit, simple form:
        const float* ar = aggr + (size_t)n * FDIM;
#pragma unroll
        for (int j = 0; j < 8; ++j) A1.h[j]     = (_Float16)ar[hi * 8 + j];        // K 32..47
#pragma unroll
        for (int j = 0; j < 8; ++j) A1.h[8 + j] = (_Float16)ar[16 + hi * 8 + j];   // K 48..63
    }

#pragma unroll
    for (int nt = 0; nt < 4; ++nt) {
        v8f c;
#pragma unroll
        for (int v = 0; v < 8; ++v) c[v] = bias1[nt];
        c = wmma_f16(A0.v, B1[0][nt].v, c);
        c = wmma_f16(A1.v, B1[1][nt].v, c);
#pragma unroll
        for (int v = 0; v < 8; ++v) {
            float r = c[v] > 0.0f ? c[v] : 0.0f;
            myLds[(v + 8 * hi) * 72 + nt * 16 + lm] = (_Float16)r;
        }
    }
    asm volatile("s_wait_dscnt 0" ::: "memory");

    const uint4* pr = (const uint4*)(myLds + lm * 72);
    FragH A2a, A2b;
    A2a.q[0] = pr[hi];     A2a.q[1] = pr[2 + hi];
    A2b.q[0] = pr[4 + hi]; A2b.q[1] = pr[6 + hi];

#pragma unroll
    for (int nt = 0; nt < 2; ++nt) {
        v8f c;
#pragma unroll
        for (int v = 0; v < 8; ++v) c[v] = bias2[nt];
        c = wmma_f16(A2a.v, B2[0][nt].v, c);
        c = wmma_f16(A2b.v, B2[1][nt].v, c);
#pragma unroll
        for (int v = 0; v < 8; ++v) {
            const int row = v + 8 * hi;
            const int g   = batch[tile * 16 + row];
            atomAddF32(&sums[g * FDIM + nt * 16 + lm], c[v]);
        }
    }
}

// ---------------------------------------------------------------------------
// Kernel 4: head MLP, one block per graph (tiny: 100 x (32->64->2)).
// ---------------------------------------------------------------------------
__global__ void head_kernel(const float* __restrict__ sums,
                            const float* __restrict__ cnt,
                            const float* __restrict__ w1, const float* __restrict__ b1,
                            const float* __restrict__ w2, const float* __restrict__ b2,
                            float* __restrict__ out) {
    __shared__ float pooled[FDIM];
    __shared__ float hid[64];
    const int g = blockIdx.x;
    const int t = threadIdx.x;
    float c = cnt[g];
    c = c < 1.0f ? 1.0f : c;
    if (t < FDIM) pooled[t] = sums[g * FDIM + t] / c;
    __syncthreads();
    float acc = b1[t];
#pragma unroll
    for (int k = 0; k < FDIM; ++k) acc += pooled[k] * w1[k * 64 + t];
    hid[t] = acc > 0.0f ? acc : 0.0f;
    __syncthreads();
    if (t < 2) {
        float o = b2[t];
#pragma unroll
        for (int k = 0; k < 64; ++k) o += hid[k] * w2[k * 2 + t];
        out[g * 2 + t] = o;
    }
}

// ---------------------------------------------------------------------------
extern "C" void kernel_launch(void* const* d_in, const int* in_sizes, int n_in,
                              void* d_out, int out_size, void* d_ws, size_t ws_size,
                              hipStream_t stream) {
    const float* x       = (const float*)d_in[0];
    const int*   ei      = (const int*)d_in[1];
    const int*   batch   = (const int*)d_in[2];
    const float* phi_w1  = (const float*)d_in[3];
    const float* phi_b1  = (const float*)d_in[4];
    const float* phi_w2  = (const float*)d_in[5];
    const float* phi_b2  = (const float*)d_in[6];
    const float* gam_w1  = (const float*)d_in[7];
    const float* gam_b1  = (const float*)d_in[8];
    const float* gam_w2  = (const float*)d_in[9];
    const float* gam_b2  = (const float*)d_in[10];
    const float* mlp_w1  = (const float*)d_in[11];
    const float* mlp_b1  = (const float*)d_in[12];
    const float* mlp_w2  = (const float*)d_in[13];
    const float* mlp_b2  = (const float*)d_in[14];
    float* out = (float*)d_out;

    const int* src = ei;            // edge_index[0]
    const int* dst = ei + N_EDGES;  // edge_index[1]

    // workspace layout (all < 12 MB)
    char* ws = (char*)d_ws;
    _Float16* xh   = (_Float16*)(ws + 0);                         // 3,200,000 B
    float*    aggr = (float*)(ws + (4u << 20));                   // 6,400,000 B
    float*    sums = (float*)(ws + (11u << 20));                  // 12,800 B
    float*    cnt  = (float*)(ws + (11u << 20) + 16384);          // 400 B

    // 0) f16 convert + zero accumulators
    init_kernel<<<6250, 256, 0, stream>>>(x, xh, aggr, sums, cnt);
    // 1) per-graph counts
    count_kernel<<<(N_NODES + 255) / 256, 256, 0, stream>>>(batch, cnt);
    // 2) edge MLP + scatter (100,000 16-edge tiles)
    edge_mlp_kernel<<<2048, 256, 0, stream>>>(xh, src, dst,
                                              phi_w1, phi_b1, phi_w2, phi_b2,
                                              aggr, N_EDGES / 16);
    // 3) node MLP + pooled scatter (3,125 16-node tiles)
    node_mlp_kernel<<<(N_NODES / 16 + 7) / 8, 256, 0, stream>>>(xh, aggr, batch,
                                                                gam_w1, gam_b1, gam_w2, gam_b2,
                                                                sums, N_NODES / 16);
    // 4) head MLP
    head_kernel<<<N_GRAPHS, 64, 0, stream>>>(sums, cnt, mlp_w1, mlp_b1, mlp_w2, mlp_b2, out);
    (void)in_sizes; (void)n_in; (void)out_size; (void)ws_size;
}
